// TransformerBlock_15753940042096
// MI455X (gfx1250) — matrix operand order repaired and checked by reading, not compiled, not run on hardware
//
#include <hip/hip_runtime.h>
#include <hip/hip_bf16.h>

// ---------------------------------------------------------------------------
// Problem constants (from reference)
// ---------------------------------------------------------------------------
#define DMODEL 1024
#define DFF    4096
#define SEQLEN 2048
#define NBATCH 4
#define NHEADS 16
#define DHEAD  64
#define NTOK   (NBATCH * SEQLEN)   // 8192 rows

typedef __bf16 bf16;
typedef __attribute__((ext_vector_type(16))) __bf16 v16bf;
typedef __attribute__((ext_vector_type(8)))  __bf16 v8bf;
typedef __attribute__((ext_vector_type(8)))  float  v8f;

#define NEG_INF (-1.0e30f)

// ---------------------------------------------------------------------------
// WMMA helpers (CDNA5: V_WMMA_F32_16X16X32_BF16, wave32)
// ---------------------------------------------------------------------------
__device__ __forceinline__ v8f v8f_zero() {
    v8f z;
#pragma unroll
    for (int i = 0; i < 8; ++i) z[i] = 0.0f;
    return z;
}

__device__ __forceinline__ v8f wmma_bf16(v16bf a, v16bf b, v8f c) {
    // 8 args: (neg_a, A, neg_b, B, c_mod, C, reuse_a, reuse_b)
    return __builtin_amdgcn_wmma_f32_16x16x32_bf16(false, a, false, b,
                                                   (short)0, c, false, false);
}

__device__ __forceinline__ float fast_rcp(float x) {
    return __builtin_amdgcn_rcpf(x);   // v_rcp_f32 (bf16-output precision is fine)
}

// A-matrix 16x32 bf16 fragment from row-major A (lda elements per row).
// ISA layout: lanes 0-15 hold row M=lane, K={k0..k0+7, k0+16..k0+23};
//             lanes 16-31 hold row M=lane-16, K={k0+8..k0+15, k0+24..k0+31}.
__device__ __forceinline__ v16bf load_a_frag(const bf16* A, int lda,
                                             int row0, int k0, int lane) {
    const int l  = lane & 15;
    const int hi = (lane >> 4) & 1;
    const bf16* p = A + (size_t)(row0 + l) * lda + k0 + hi * 8;
    v8bf lo = *(const v8bf*)p;
    v8bf hh = *(const v8bf*)(p + 16);
    v16bf r;
#pragma unroll
    for (int i = 0; i < 8; ++i) { r[i] = lo[i]; r[i + 8] = hh[i]; }
    return r;
}

// B-matrix 32x16 bf16 fragment from transposed storage Bt[N][K] (ldb = K).
// ISA layout: lanes 0-15 hold col N=lane, K={k0..k0+7, k0+16..k0+23};
//             lanes 16-31 hold col N=lane-16, K={k0+8..k0+15, k0+24..k0+31}.
__device__ __forceinline__ v16bf load_b_frag(const bf16* Bt, int ldb,
                                             int col0, int k0, int lane) {
    const int n  = lane & 15;
    const int hi = (lane >> 4) & 1;
    const bf16* p = Bt + (size_t)(col0 + n) * ldb + k0 + hi * 8;
    v8bf lo = *(const v8bf*)p;
    v8bf hh = *(const v8bf*)(p + 16);
    v16bf r;
#pragma unroll
    for (int i = 0; i < 8; ++i) { r[i] = lo[i]; r[i + 8] = hh[i]; }
    return r;
}

// ---------------------------------------------------------------------------
// Elementwise / prep kernels
// ---------------------------------------------------------------------------

// Wt[n*K + k] = (bf16) W[k*N + n]   (cast + transpose weights once)
__global__ void cast_transpose_kernel(const float* __restrict__ W,
                                      bf16* __restrict__ Wt, int K, int N) {
    int kk = blockIdx.x * 256 + threadIdx.x;
    int n  = blockIdx.y;
    if (kk < K) Wt[(size_t)n * K + kk] = (bf16)W[(size_t)kk * N + n];
}

// RMSNorm over last dim (1024) of fp32 input, bf16 output. One block per row.
__global__ void rmsnorm_cast_kernel(const float* __restrict__ x,
                                    const float* __restrict__ g,
                                    bf16* __restrict__ out) {
    const int row = blockIdx.x;
    const float* xr = x + (size_t)row * DMODEL;
    __shared__ float red[8];
    float ss = 0.0f;
    for (int i = threadIdx.x; i < DMODEL; i += 256) {
        float v = xr[i];
        ss += v * v;
    }
#pragma unroll
    for (int off = 16; off > 0; off >>= 1) ss += __shfl_xor(ss, off, 32);
    const int wv = threadIdx.x >> 5;
    if ((threadIdx.x & 31) == 0) red[wv] = ss;
    __syncthreads();
    float tot = 0.0f;
#pragma unroll
    for (int i = 0; i < 8; ++i) tot += red[i];
    const float rinv = rsqrtf(tot * (1.0f / DMODEL) + 1e-5f);
    bf16* orow = out + (size_t)row * DMODEL;
    for (int i = threadIdx.x; i < DMODEL; i += 256)
        orow[i] = (bf16)(xr[i] * rinv * g[i]);
}

// RoPE applied in-place to q and k (bf16, layout [B*S, H*DH])
__global__ void rope_kernel(bf16* __restrict__ q, bf16* __restrict__ k,
                            const int* __restrict__ tp) {
    int idx = blockIdx.x * 256 + threadIdx.x;      // B*S*H*32 pairs
    int i    = idx & 31;  idx >>= 5;
    int h    = idx & 15;  idx >>= 4;
    int s    = idx & (SEQLEN - 1);
    int b    = idx >> 11;
    float pos  = (float)tp[s];
    float freq = __expf((-2.0f * (float)i / (float)DHEAD) * __logf(10000.0f));
    float ang  = pos * freq;
    float c = __cosf(ang), sn = __sinf(ang);
    size_t off = ((size_t)(b * SEQLEN + s)) * DMODEL + h * DHEAD + 2 * i;
    float q1 = (float)q[off], q2 = (float)q[off + 1];
    q[off]     = (bf16)(q1 * c - q2 * sn);
    q[off + 1] = (bf16)(q1 * sn + q2 * c);
    float k1 = (float)k[off], k2 = (float)k[off + 1];
    k[off]     = (bf16)(k1 * c - k2 * sn);
    k[off + 1] = (bf16)(k1 * sn + k2 * c);
}

// vt[(b*H+h)*DH + d][s] = v[(b*S+s)*D + h*DH + d]  (per-head V transpose)
__global__ void transpose_v_kernel(const bf16* __restrict__ v,
                                   bf16* __restrict__ vt) {
    int s   = blockIdx.x * 256 + threadIdx.x;
    int row = blockIdx.y;                 // 0 .. B*H*DH-1
    int d   = row & (DHEAD - 1);
    int bh  = row >> 6;
    int h   = bh & (NHEADS - 1);
    int b   = bh >> 4;
    vt[(size_t)row * SEQLEN + s] =
        v[((size_t)(b * SEQLEN + s)) * DMODEL + h * DHEAD + d];
}

// ---------------------------------------------------------------------------
// GEMM: C[M,N] = A[M,K](bf16,row-major) x Bt[N,K](bf16, transposed weights)
// Block = 256 threads = 8 waves arranged 2(M) x 4(N); wave tile 64x64
// (16 WMMAs per 8 fragment loads). Block tile 128x256.
// EPI 0: store bf16. EPI 1: fp32 out = resid + acc.
// ---------------------------------------------------------------------------
template <int EPI>
__global__ void gemm_wmma_kernel(const bf16* __restrict__ A,
                                 const bf16* __restrict__ Bt,
                                 const float* __restrict__ resid,
                                 void* __restrict__ Cout,
                                 int M, int N, int K) {
    const int lane = threadIdx.x & 31;
    const int w    = threadIdx.x >> 5;
    const int wm   = w >> 2;         // 0..1
    const int wn   = w & 3;          // 0..3
    const int wr0  = blockIdx.y * 128 + wm * 64;
    const int wc0  = blockIdx.x * 256 + wn * 64;

    v8f acc[4][4];
#pragma unroll
    for (int i = 0; i < 4; ++i)
#pragma unroll
        for (int j = 0; j < 4; ++j) acc[i][j] = v8f_zero();

    for (int k0 = 0; k0 < K; k0 += 32) {
        __builtin_prefetch(A + (size_t)(wr0 + (lane & 15)) * K + k0 + 64, 0, 1);
        v16bf a[4], b[4];
#pragma unroll
        for (int i = 0; i < 4; ++i)
            a[i] = load_a_frag(A, K, wr0 + 16 * i, k0, lane);
#pragma unroll
        for (int j = 0; j < 4; ++j)
            b[j] = load_b_frag(Bt, K, wc0 + 16 * j, k0, lane);
#pragma unroll
        for (int i = 0; i < 4; ++i)
#pragma unroll
            for (int j = 0; j < 4; ++j)
                acc[i][j] = wmma_bf16(a[i], b[j], acc[i][j]);
    }

    const int n  = lane & 15;
    const int mo = (lane & 16) ? 8 : 0;
#pragma unroll
    for (int i = 0; i < 4; ++i) {
#pragma unroll
        for (int j = 0; j < 4; ++j) {
#pragma unroll
            for (int jj = 0; jj < 8; ++jj) {
                size_t r = (size_t)(wr0 + 16 * i + mo + jj);
                size_t c = (size_t)(wc0 + 16 * j + n);
                if (EPI == 0) {
                    ((bf16*)Cout)[r * N + c] = (bf16)acc[i][j][jj];
                } else {
                    ((float*)Cout)[r * N + c] = resid[r * N + c] + acc[i][j][jj];
                }
            }
        }
    }
}

// ---------------------------------------------------------------------------
// Fused SwiGLU dual GEMM: G = silu(A@W1) * (A@W3), bf16 out.
// Block = 8 waves: 2(M) x 4(N); wave tile 32x32 per matrix. Block tile 64x128.
// ---------------------------------------------------------------------------
__global__ void gemm_dual_silu_kernel(const bf16* __restrict__ A,
                                      const bf16* __restrict__ B1t,
                                      const bf16* __restrict__ B3t,
                                      bf16* __restrict__ G,
                                      int M, int N, int K) {
    const int lane = threadIdx.x & 31;
    const int w    = threadIdx.x >> 5;
    const int wm   = w >> 2;
    const int wn   = w & 3;
    const int wr0  = blockIdx.y * 64 + wm * 32;
    const int wc0  = blockIdx.x * 128 + wn * 32;

    v8f acc1[2][2], acc3[2][2];
#pragma unroll
    for (int i = 0; i < 2; ++i)
#pragma unroll
        for (int j = 0; j < 2; ++j) { acc1[i][j] = v8f_zero(); acc3[i][j] = v8f_zero(); }

    for (int k0 = 0; k0 < K; k0 += 32) {
        v16bf a[2], b1[2], b3[2];
#pragma unroll
        for (int i = 0; i < 2; ++i)
            a[i] = load_a_frag(A, K, wr0 + 16 * i, k0, lane);
#pragma unroll
        for (int j = 0; j < 2; ++j) {
            b1[j] = load_b_frag(B1t, K, wc0 + 16 * j, k0, lane);
            b3[j] = load_b_frag(B3t, K, wc0 + 16 * j, k0, lane);
        }
#pragma unroll
        for (int i = 0; i < 2; ++i)
#pragma unroll
            for (int j = 0; j < 2; ++j) {
                acc1[i][j] = wmma_bf16(a[i], b1[j], acc1[i][j]);
                acc3[i][j] = wmma_bf16(a[i], b3[j], acc3[i][j]);
            }
    }

    const int n  = lane & 15;
    const int mo = (lane & 16) ? 8 : 0;
#pragma unroll
    for (int i = 0; i < 2; ++i)
#pragma unroll
        for (int j = 0; j < 2; ++j)
#pragma unroll
            for (int jj = 0; jj < 8; ++jj) {
                size_t r = (size_t)(wr0 + 16 * i + mo + jj);
                size_t c = (size_t)(wc0 + 16 * j + n);
                float u = acc1[i][j][jj];
                float v = acc3[i][j][jj];
                // silu(u)*v with v_rcp_f32 instead of IEEE divide
                float sig  = fast_rcp(1.0f + __expf(-u));
                G[r * N + c] = (bf16)(u * sig * v);
            }
}

// ---------------------------------------------------------------------------
// Causal flash attention. One wave per (batch, head, 16-query tile).
// Block = 128 threads (4 waves). QK^T and PV via WMMA bf16; softmax fp32.
// ---------------------------------------------------------------------------
__global__ void attn_kernel(const bf16* __restrict__ q,
                            const bf16* __restrict__ k,
                            const bf16* __restrict__ vt,
                            bf16* __restrict__ ctxOut) {
    __shared__ __bf16 plds[4][16][32];   // per-wave P staging (transpose via LDS)

    const int lane = threadIdx.x & 31;
    const int wv   = threadIdx.x >> 5;
    const int wid  = blockIdx.x * 4 + wv;          // 0 .. B*H*(S/16)-1
    const int qt   = wid & 127;                    // S/16 = 128 q-tiles
    const int h    = (wid >> 7) & (NHEADS - 1);
    const int b    = wid >> 11;
    const int q0   = qt * 16;
    const int hoff = h * DHEAD;

    const bf16* Qb  = q  + (size_t)b * SEQLEN * DMODEL;
    const bf16* Kb  = k  + (size_t)b * SEQLEN * DMODEL;
    const bf16* Vtb = vt + (size_t)(b * NHEADS + h) * DHEAD * SEQLEN;

    // Q fragments: 16 rows x 64 dh = 2 A-fragments, loaded once.
    v16bf qa[2];
#pragma unroll
    for (int t = 0; t < 2; ++t)
        qa[t] = load_a_frag(Qb, DMODEL, q0, hoff + 32 * t, lane);

    v8f ctx[4];
#pragma unroll
    for (int t = 0; t < 4; ++t) ctx[t] = v8f_zero();
    float m[8], l[8];
#pragma unroll
    for (int j = 0; j < 8; ++j) { m[j] = NEG_INF; l[j] = 0.0f; }

    const int n  = lane & 15;
    const int mo = (lane & 16) ? 8 : 0;
    const int kend = q0 + 16;                      // exclusive causal key bound

    for (int kt = 0; kt < kend; kt += 32) {
        // --- scores for 32 keys: two 16x16 tiles, K-dim = dh (64) ---
        v8f sa = v8f_zero(), sb = v8f_zero();
#pragma unroll
        for (int t = 0; t < 2; ++t) {
            v16bf kb0 = load_b_frag(Kb, DMODEL, kt,      hoff + 32 * t, lane);
            v16bf kb1 = load_b_frag(Kb, DMODEL, kt + 16, hoff + 32 * t, lane);
            sa = wmma_bf16(qa[t], kb0, sa);
            sb = wmma_bf16(qa[t], kb1, sb);
        }

        // --- masked online softmax (rows spread across 16-lane halves) ---
        float pa[8], pb[8];
#pragma unroll
        for (int j = 0; j < 8; ++j) {
            int qrow = q0 + mo + j;
            float s1 = sa[j] * 0.125f; if (kt + n > qrow)      s1 = NEG_INF;
            float s2 = sb[j] * 0.125f; if (kt + 16 + n > qrow) s2 = NEG_INF;
            float r = fmaxf(s1, s2);
#pragma unroll
            for (int off = 8; off > 0; off >>= 1)
                r = fmaxf(r, __shfl_xor(r, off, 32));
            float nm = fmaxf(m[j], r);
            float al = __expf(m[j] - nm);
            float p1 = __expf(s1 - nm);
            float p2 = __expf(s2 - nm);
            float rs = p1 + p2;
#pragma unroll
            for (int off = 8; off > 0; off >>= 1)
                rs += __shfl_xor(rs, off, 32);
            l[j] = l[j] * al + rs;
            m[j] = nm;
#pragma unroll
            for (int t = 0; t < 4; ++t) ctx[t][j] *= al;
            pa[j] = p1; pb[j] = p2;
        }

        // --- transpose P (C-layout -> A-layout) through LDS ---
#pragma unroll
        for (int j = 0; j < 8; ++j) {
            plds[wv][mo + j][n]      = (bf16)pa[j];
            plds[wv][mo + j][16 + n] = (bf16)pb[j];
        }
        v16bf pf = load_a_frag((const bf16*)&plds[wv][0][0], 32, 0, 0, lane);

        // --- ctx += P(16x32) x V(32x64), V pre-transposed [dh][s] ---
#pragma unroll
        for (int t = 0; t < 4; ++t) {
            v16bf vf = load_b_frag(Vtb, SEQLEN, t * 16, kt, lane);
            ctx[t] = wmma_bf16(pf, vf, ctx[t]);
        }
    }

    // --- normalize and store ---
#pragma unroll
    for (int j = 0; j < 8; ++j) {
        float inv = fast_rcp(l[j]);
        size_t row = (size_t)(b * SEQLEN + q0 + mo + j);
#pragma unroll
        for (int t = 0; t < 4; ++t)
            ctxOut[row * DMODEL + hoff + t * 16 + n] = (bf16)(ctx[t][j] * inv);
    }
}

// ---------------------------------------------------------------------------
// Host-side orchestration
// ---------------------------------------------------------------------------
extern "C" void kernel_launch(void* const* d_in, const int* in_sizes, int n_in,
                              void* d_out, int out_size, void* d_ws, size_t ws_size,
                              hipStream_t stream) {
    (void)in_sizes; (void)n_in; (void)out_size; (void)ws_size;

    const float* x  = (const float*)d_in[0];
    const int*   tp = (const int*)d_in[1];
    const float* g1 = (const float*)d_in[2];
    const float* g2 = (const float*)d_in[3];
    const float* wq = (const float*)d_in[4];
    const float* wk = (const float*)d_in[5];
    const float* wvw = (const float*)d_in[6];
    const float* wo = (const float*)d_in[7];
    const float* w1 = (const float*)d_in[8];
    const float* w2 = (const float*)d_in[9];
    const float* w3 = (const float*)d_in[10];
    float* out = (float*)d_out;

    // workspace carve-up (256B aligned)
    char* p = (char*)d_ws;
    auto alloc = [&](size_t bytes) -> char* {
        char* r = p;
        p += (bytes + 255) & ~(size_t)255;
        return r;
    };
    bf16* wqT = (bf16*)alloc((size_t)DMODEL * DMODEL * 2);
    bf16* wkT = (bf16*)alloc((size_t)DMODEL * DMODEL * 2);
    bf16* wvT = (bf16*)alloc((size_t)DMODEL * DMODEL * 2);
    bf16* woT = (bf16*)alloc((size_t)DMODEL * DMODEL * 2);
    bf16* w1T = (bf16*)alloc((size_t)DMODEL * DFF * 2);
    bf16* w3T = (bf16*)alloc((size_t)DMODEL * DFF * 2);
    bf16* w2T = (bf16*)alloc((size_t)DFF * DMODEL * 2);
    bf16* hb   = (bf16*)alloc((size_t)NTOK * DMODEL * 2);
    bf16* qb   = (bf16*)alloc((size_t)NTOK * DMODEL * 2);
    bf16* kb   = (bf16*)alloc((size_t)NTOK * DMODEL * 2);
    bf16* vb   = (bf16*)alloc((size_t)NTOK * DMODEL * 2);
    bf16* vtb  = (bf16*)alloc((size_t)NTOK * DMODEL * 2);
    bf16* ctxb = (bf16*)alloc((size_t)NTOK * DMODEL * 2);
    float* yb  = (float*)alloc((size_t)NTOK * DMODEL * 4);
    bf16* h2b  = (bf16*)alloc((size_t)NTOK * DMODEL * 2);
    bf16* gb   = (bf16*)alloc((size_t)NTOK * DFF * 2);

    // 1) weight cast+transpose
    cast_transpose_kernel<<<dim3(DMODEL / 256, DMODEL), 256, 0, stream>>>(wq, wqT, DMODEL, DMODEL);
    cast_transpose_kernel<<<dim3(DMODEL / 256, DMODEL), 256, 0, stream>>>(wk, wkT, DMODEL, DMODEL);
    cast_transpose_kernel<<<dim3(DMODEL / 256, DMODEL), 256, 0, stream>>>(wvw, wvT, DMODEL, DMODEL);
    cast_transpose_kernel<<<dim3(DMODEL / 256, DMODEL), 256, 0, stream>>>(wo, woT, DMODEL, DMODEL);
    cast_transpose_kernel<<<dim3(DMODEL / 256, DFF), 256, 0, stream>>>(w1, w1T, DMODEL, DFF);
    cast_transpose_kernel<<<dim3(DMODEL / 256, DFF), 256, 0, stream>>>(w3, w3T, DMODEL, DFF);
    cast_transpose_kernel<<<dim3(DFF / 256, DMODEL), 256, 0, stream>>>(w2, w2T, DFF, DMODEL);

    // 2) h = rmsnorm(x, g1) in bf16
    rmsnorm_cast_kernel<<<NTOK, 256, 0, stream>>>(x, g1, hb);

    // 3) q/k/v projections (bf16 out)  — block tile 128x256
    dim3 ggrid(DMODEL / 256, NTOK / 128);
    gemm_wmma_kernel<0><<<ggrid, 256, 0, stream>>>(hb, wqT, nullptr, qb, NTOK, DMODEL, DMODEL);
    gemm_wmma_kernel<0><<<ggrid, 256, 0, stream>>>(hb, wkT, nullptr, kb, NTOK, DMODEL, DMODEL);
    gemm_wmma_kernel<0><<<ggrid, 256, 0, stream>>>(hb, wvT, nullptr, vb, NTOK, DMODEL, DMODEL);

    // 4) RoPE on q,k
    rope_kernel<<<(NBATCH * SEQLEN * NHEADS * 32) / 256, 256, 0, stream>>>(qb, kb, tp);

    // 5) transpose V per head for PV WMMA B-fragments
    transpose_v_kernel<<<dim3(SEQLEN / 256, NBATCH * NHEADS * DHEAD), 256, 0, stream>>>(vb, vtb);

    // 6) flash attention
    attn_kernel<<<(NBATCH * NHEADS * (SEQLEN / 16)) / 4, 128, 0, stream>>>(qb, kb, vtb, ctxb);

    // 7) y = x + ctx @ wo  (fp32 residual out)
    gemm_wmma_kernel<1><<<ggrid, 256, 0, stream>>>(ctxb, woT, x, yb, NTOK, DMODEL, DMODEL);

    // 8) h2 = rmsnorm(y, g2)
    rmsnorm_cast_kernel<<<NTOK, 256, 0, stream>>>(yb, g2, h2b);

    // 9) gate = silu(h2@w1) * (h2@w3)  (fused dual GEMM)
    gemm_dual_silu_kernel<<<dim3(DFF / 128, NTOK / 64), 256, 0, stream>>>(
        h2b, w1T, w3T, gb, NTOK, DFF, DMODEL);

    // 10) out = y + gate @ w2
    gemm_wmma_kernel<1><<<ggrid, 256, 0, stream>>>(gb, w2T, yb, out, NTOK, DMODEL, DFF);
}
